// MultiHeadAttention_30477087932440
// MI455X (gfx1250) — compile-verified, matrix operand order checked by reading, not run on hardware
//
#include <hip/hip_runtime.h>
#include <hip/hip_bf16.h>

// ---------------------------------------------------------------------------
// MI455X (gfx1250) multi-head attention, f16 WMMA path with f32 accumulate.
//   B=8, C=512, H=W=32 -> N=1024, heads=8, head_dim=64
// Pipeline:
//   0) one-time fp32 -> f16 conversion of x and all W matrices (bandwidth-bound)
//   1) proj kernel (x3): Y = W x + b      (Q,K i-major f16; V o-major f16)
//   2) flash attention per (b,h,64-query block), 64-key steps -> ctx (i-major)
//   3) out-proj: out = Wo ctx + bo + x    (fp32)
// Workspace (f16): Q,K,V,ctx,xh (4M elems each) + 4x W (256K each) = ~44MB.
// ---------------------------------------------------------------------------

typedef __attribute__((ext_vector_type(16))) _Float16 v16h;
typedef __attribute__((ext_vector_type(8)))  _Float16 v8h;
typedef __attribute__((ext_vector_type(4)))  _Float16 v4h;
typedef __attribute__((ext_vector_type(8)))  float    v8f;

#define B_    8
#define C_    512
#define NPOS  1024
#define NH    8
#define HD    64

// ----- WMMA fragment helpers (wave32, 16x16x32 f16) -------------------------
// A 16x32 per-lane layout: lane holds row m=lane&15; element e maps to
// K = (e<8 ? e : e+8) + 8*(lane>=16)  -> two contiguous 16B runs per lane.
// B 32x16 mirrors this with lane holding column n=lane&15.

__device__ __forceinline__ v16h frag_from_h16(const _Float16* row, int hlf) {
  const v8h lo = *(const v8h*)(row + hlf * 8);
  const v8h hi = *(const v8h*)(row + hlf * 8 + 16);
  v16h r;
#pragma unroll
  for (int e = 0; e < 8; ++e) { r[e] = lo[e]; r[e + 8] = hi[e]; }
  return r;
}

__device__ __forceinline__ v8f wmma16(v16h a, v16h b, v8f c) {
  return __builtin_amdgcn_wmma_f32_16x16x32_f16(
      /*neg_a=*/false, a, /*neg_b=*/false, b,
      /*c_mod=*/(short)0, c, /*reuse_a=*/false, /*reuse_b=*/false);
}

// ----- 0) one-time fp32 -> f16 conversion ----------------------------------
__global__ __launch_bounds__(256) void cvt_f32_to_f16(
    const float* __restrict__ src, _Float16* __restrict__ dst, int n4) {
  const int i = blockIdx.x * 256 + threadIdx.x;
  if (i < n4) {
    const float4 f = ((const float4*)src)[i];
    v4h h;
    h[0] = (_Float16)f.x; h[1] = (_Float16)f.y;
    h[2] = (_Float16)f.z; h[3] = (_Float16)f.w;
    ((v4h*)dst)[i] = h;
  }
}

// ----- 1) projection kernel: Y[o,n] = sum_c W[o,c] X[c,n] + bias[o] --------
// All-f16 inputs: hot loop is pure b128 load -> v_wmma, no conversions.
// MODE 0: store f16 o-major  buf[b][o][n]   (V)
// MODE 1: store f16 i-major  buf[b][n][o]   (Q scaled by 1/8, K)
template <int MODE>
__global__ __launch_bounds__(256) void proj_kernel(
    const _Float16* __restrict__ xh, const _Float16* __restrict__ Wh,
    const float* __restrict__ bias, _Float16* __restrict__ out, float scale) {
  __shared__ _Float16 Xt[128][40];  // X tile transposed: [n][k], pad 32->40

  const int b   = blockIdx.z;
  const int m0  = blockIdx.y * 64;
  const int n0  = blockIdx.x * 128;
  const int tid = threadIdx.x;
  const int lane = tid & 31;
  const int wv   = tid >> 5;
  const int wm   = wv >> 2;    // 0..1  (M sub-tile, 32 rows)
  const int wn   = wv & 3;     // 0..3  (N sub-tile, 32 cols)
  const int hlf  = lane >> 4;
  const int cn   = lane & 15;

  const _Float16* xb = xh + (size_t)b * C_ * NPOS;

  v8f zero = {};
  v8f acc[2][2];
#pragma unroll
  for (int i = 0; i < 2; ++i)
#pragma unroll
    for (int j = 0; j < 2; ++j) acc[i][j] = zero;

  const int sk = tid >> 3;         // 0..31 : k row staged by this thread
  const int sn = (tid & 7) * 16;   // 0..112: first of 16 n columns

  for (int k0 = 0; k0 < C_; k0 += 32) {
    __syncthreads();
    // stage 32(K) x 128(N) f16 tile of X transposed into LDS
    const _Float16* src = xb + (size_t)(k0 + sk) * NPOS + n0 + sn;
    const v8h h0 = *(const v8h*)(src + 0);
    const v8h h1 = *(const v8h*)(src + 8);
#pragma unroll
    for (int e = 0; e < 8; ++e) {
      Xt[sn + e][sk]     = h0[e];
      Xt[sn + 8 + e][sk] = h1[e];
    }
    __syncthreads();

    v16h bf[2];
#pragma unroll
    for (int ni = 0; ni < 2; ++ni)
      bf[ni] = frag_from_h16(&Xt[wn * 32 + ni * 16 + cn][0], hlf);
#pragma unroll
    for (int mi = 0; mi < 2; ++mi) {
      const int m = m0 + wm * 32 + mi * 16 + cn;
      const v16h a = frag_from_h16(Wh + (size_t)m * C_ + k0, hlf);
#pragma unroll
      for (int ni = 0; ni < 2; ++ni)
        acc[mi][ni] = wmma16(a, bf[ni], acc[mi][ni]);
    }
  }

#pragma unroll
  for (int mi = 0; mi < 2; ++mi)
#pragma unroll
    for (int ni = 0; ni < 2; ++ni)
#pragma unroll
      for (int r = 0; r < 8; ++r) {
        const int o = m0 + wm * 32 + mi * 16 + r + 8 * hlf;
        const int n = n0 + wn * 32 + ni * 16 + cn;
        const float v = (acc[mi][ni][r] + bias[o]) * scale;
        if (MODE == 0)
          out[((size_t)b * C_ + o) * NPOS + n] = (_Float16)v;
        else
          out[((size_t)b * NPOS + n) * C_ + o] = (_Float16)v;
      }
}

// ----- 2) flash attention per (b, h, 64-query block) ------------------------
// Q,K i-major [b][i][c]; V o-major [b][c][j]; ctx i-major [b][i][c].
// 4 waves, each owns 16 query rows; j-loop of 64 keys:
//   S(16x64) = Qi^T Kj (8 WMMA), online softmax, O += P Vj^T (8 WMMA).
// 64-key steps amortize the cross-lane softmax reductions over 2x more keys.
__global__ __launch_bounds__(128) void attn_kernel(
    const _Float16* __restrict__ Q, const _Float16* __restrict__ K,
    const _Float16* __restrict__ V, _Float16* __restrict__ ctx) {
  __shared__ _Float16 P[4][16][72];  // per-wave P tile (16x64, pad 64->72)

  const int ib  = blockIdx.x;  // 0..15 query block
  const int h   = blockIdx.y;
  const int b   = blockIdx.z;
  const int tid = threadIdx.x;
  const int lane = tid & 31;
  const int wv   = tid >> 5;   // 0..3
  const int hlf  = lane >> 4;
  const int cn   = lane & 15;
  const int i0   = ib * 64 + wv * 16;

  const _Float16* Qb = Q + (size_t)b * NPOS * C_ + h * HD;
  const _Float16* Kb = K + (size_t)b * NPOS * C_ + h * HD;
  const _Float16* Vb = V + ((size_t)b * C_ + h * HD) * NPOS;

  v16h aq[2];
#pragma unroll
  for (int kk = 0; kk < 2; ++kk)
    aq[kk] = frag_from_h16(Qb + (size_t)(i0 + cn) * C_ + kk * 32, hlf);

  v8f zero = {};
  v8f o[4];
#pragma unroll
  for (int t = 0; t < 4; ++t) o[t] = zero;
  float mrun[8], lrun[8];
#pragma unroll
  for (int r = 0; r < 8; ++r) { mrun[r] = -3.0e38f; lrun[r] = 0.0f; }

  for (int j0 = 0; j0 < NPOS; j0 += 64) {
    // --- scores S(16x64) = Q^T K (scale folded into Q) ---
    v8f s[4];
#pragma unroll
    for (int jj = 0; jj < 4; ++jj) s[jj] = zero;
#pragma unroll
    for (int jj = 0; jj < 4; ++jj)
#pragma unroll
      for (int kk = 0; kk < 2; ++kk) {
        const v16h kf =
            frag_from_h16(Kb + (size_t)(j0 + jj * 16 + cn) * C_ + kk * 32, hlf);
        s[jj] = wmma16(aq[kk], kf, s[jj]);
      }
    // prefetch next key block's row for this lane (global_prefetch_b8)
    if (j0 + 64 < NPOS)
      __builtin_prefetch(Kb + (size_t)(j0 + 64 + cn) * C_, 0, 3);

    // --- online softmax; row m = r + 8*hlf lives across 16 lanes of a half ---
#pragma unroll
    for (int r = 0; r < 8; ++r) {
      float ms = fmaxf(fmaxf(s[0][r], s[1][r]), fmaxf(s[2][r], s[3][r]));
#pragma unroll
      for (int msk = 1; msk < 16; msk <<= 1)
        ms = fmaxf(ms, __shfl_xor(ms, msk, 32));
      const float mn = fmaxf(mrun[r], ms);
      const float cr = __expf(mrun[r] - mn);
      float p[4], ps = 0.0f;
#pragma unroll
      for (int q = 0; q < 4; ++q) { p[q] = __expf(s[q][r] - mn); ps += p[q]; }
#pragma unroll
      for (int msk = 1; msk < 16; msk <<= 1)
        ps += __shfl_xor(ps, msk, 32);
      lrun[r] = lrun[r] * cr + ps;
      mrun[r] = mn;
#pragma unroll
      for (int t = 0; t < 4; ++t) o[t][r] *= cr;
#pragma unroll
      for (int q = 0; q < 4; ++q)
        P[wv][r + 8 * hlf][cn + 16 * q] = (_Float16)p[q];
    }
    __builtin_amdgcn_wave_barrier();                 // keep compiler order
    asm volatile("s_wait_dscnt 0" ::: "memory");     // LDS writes visible

    // --- P (D-layout) reloaded as A fragments, O += P * V^T ---
    const v16h ap0 = frag_from_h16(&P[wv][cn][0], hlf);
    const v16h ap1 = frag_from_h16(&P[wv][cn][32], hlf);
#pragma unroll
    for (int t = 0; t < 4; ++t) {
      const _Float16* vrow = Vb + (size_t)(t * 16 + cn) * NPOS + j0;
      o[t] = wmma16(ap0, frag_from_h16(vrow, hlf), o[t]);
      o[t] = wmma16(ap1, frag_from_h16(vrow + 32, hlf), o[t]);
    }
  }

  // epilogue: normalize and store ctx i-major (contiguous in d across lanes)
  _Float16* cb = ctx + (size_t)b * NPOS * C_ + h * HD;
#pragma unroll
  for (int r = 0; r < 8; ++r) {
    const float inv = 1.0f / lrun[r];
    const int i = i0 + r + 8 * hlf;
#pragma unroll
    for (int t = 0; t < 4; ++t)
      cb[(size_t)i * C_ + t * 16 + cn] = (_Float16)(o[t][r] * inv);
  }
}

// ----- 3) out projection: out[o,n] = Wo ctx + bo + x ------------------------
__global__ __launch_bounds__(256) void out_proj_kernel(
    const _Float16* __restrict__ ctx, const _Float16* __restrict__ Wh,
    const float* __restrict__ bias, const float* __restrict__ x,
    float* __restrict__ out) {
  const int b   = blockIdx.z;
  const int m0  = blockIdx.y * 64;
  const int n0  = blockIdx.x * 128;
  const int tid = threadIdx.x;
  const int lane = tid & 31;
  const int wv   = tid >> 5;
  const int wm   = wv >> 2;
  const int wn   = wv & 3;
  const int hlf  = lane >> 4;
  const int cn   = lane & 15;

  const _Float16* cb = ctx + (size_t)b * NPOS * C_;  // i-major [n][c]

  v8f zero = {};
  v8f acc[2][2];
#pragma unroll
  for (int i = 0; i < 2; ++i)
#pragma unroll
    for (int j = 0; j < 2; ++j) acc[i][j] = zero;

  for (int k0 = 0; k0 < C_; k0 += 32) {
    v16h bf[2];
#pragma unroll
    for (int ni = 0; ni < 2; ++ni) {
      const int n = n0 + wn * 32 + ni * 16 + cn;
      bf[ni] = frag_from_h16(cb + (size_t)n * C_ + k0, hlf);
    }
#pragma unroll
    for (int mi = 0; mi < 2; ++mi) {
      const int m = m0 + wm * 32 + mi * 16 + cn;
      const v16h a = frag_from_h16(Wh + (size_t)m * C_ + k0, hlf);
#pragma unroll
      for (int ni = 0; ni < 2; ++ni)
        acc[mi][ni] = wmma16(a, bf[ni], acc[mi][ni]);
    }
  }

#pragma unroll
  for (int mi = 0; mi < 2; ++mi)
#pragma unroll
    for (int ni = 0; ni < 2; ++ni)
#pragma unroll
      for (int r = 0; r < 8; ++r) {
        const int o = m0 + wm * 32 + mi * 16 + r + 8 * hlf;
        const int n = n0 + wn * 32 + ni * 16 + cn;
        const size_t idx = ((size_t)b * C_ + o) * NPOS + n;
        out[idx] = acc[mi][ni][r] + bias[o] + x[idx];
      }
}

// ---------------------------------------------------------------------------
extern "C" void kernel_launch(void* const* d_in, const int* in_sizes, int n_in,
                              void* d_out, int out_size, void* d_ws, size_t ws_size,
                              hipStream_t stream) {
  (void)in_sizes; (void)n_in; (void)out_size; (void)ws_size;
  const float* x  = (const float*)d_in[0];
  const float* Wq = (const float*)d_in[1];
  const float* bq = (const float*)d_in[2];
  const float* Wk = (const float*)d_in[3];
  const float* bk = (const float*)d_in[4];
  const float* Wv = (const float*)d_in[5];
  const float* bv = (const float*)d_in[6];
  const float* Wo = (const float*)d_in[7];
  const float* bo = (const float*)d_in[8];
  float* out = (float*)d_out;

  const size_t elems = (size_t)B_ * NPOS * C_;  // 4M f16 each
  const size_t welems = (size_t)C_ * C_;        // 256K f16 each
  _Float16* Qbuf = (_Float16*)d_ws;
  _Float16* Kbuf = Qbuf + elems;
  _Float16* Vbuf = Kbuf + elems;
  _Float16* Cbuf = Vbuf + elems;
  _Float16* xhh  = Cbuf + elems;
  _Float16* Wqh  = xhh + elems;
  _Float16* Wkh  = Wqh + welems;
  _Float16* Wvh  = Wkh + welems;
  _Float16* Woh  = Wvh + welems;

  // one-time conversions (memory-bound, trivial at 23 TB/s)
  {
    const int nx4 = (int)(elems / 4), nw4 = (int)(welems / 4);
    cvt_f32_to_f16<<<(nx4 + 255) / 256, 256, 0, stream>>>(x, xhh, nx4);
    cvt_f32_to_f16<<<(nw4 + 255) / 256, 256, 0, stream>>>(Wq, Wqh, nw4);
    cvt_f32_to_f16<<<(nw4 + 255) / 256, 256, 0, stream>>>(Wk, Wkh, nw4);
    cvt_f32_to_f16<<<(nw4 + 255) / 256, 256, 0, stream>>>(Wv, Wvh, nw4);
    cvt_f32_to_f16<<<(nw4 + 255) / 256, 256, 0, stream>>>(Wo, Woh, nw4);
  }

  dim3 gp(NPOS / 128, C_ / 64, B_);
  proj_kernel<1><<<gp, 256, 0, stream>>>(xhh, Wqh, bq, Qbuf, 0.125f);  // /sqrt(hd)
  proj_kernel<1><<<gp, 256, 0, stream>>>(xhh, Wkh, bk, Kbuf, 1.0f);
  proj_kernel<0><<<gp, 256, 0, stream>>>(xhh, Wvh, bv, Vbuf, 1.0f);
  attn_kernel<<<dim3(NPOS / 64, NH, B_), 128, 0, stream>>>(Qbuf, Kbuf, Vbuf, Cbuf);
  out_proj_kernel<<<gp, 256, 0, stream>>>(Cbuf, Woh, bo, x, out);
}
